// CasualSelfAttention_50861002719792
// MI455X (gfx1250) — matrix operand order repaired and checked by reading, not compiled, not run on hardware
//
#include <hip/hip_runtime.h>
#include <hip/hip_bf16.h>

typedef __attribute__((ext_vector_type(16))) __bf16 v16bf;
typedef __attribute__((ext_vector_type(8)))  float  v8f;
typedef unsigned short u16;

union BFrag {
  v16bf v;
  u16   u[16];
  uint4 q[2];
};

__device__ __forceinline__ u16 f32_to_bf16(float f) {
  unsigned int x = __float_as_uint(f);
  x += 0x7FFFu + ((x >> 16) & 1u);   // round-to-nearest-even
  return (u16)(x >> 16);
}

#define NEG_BIG (-1.0e30f)

// Async global->LDS copy (CDNA5, tracked with ASYNCcnt). LDS byte address =
// low 32 bits of the flat address of a __shared__ pointer (ISA 10.2: LDS
// aperture addr[31:0] is the LDS offset).
__device__ __forceinline__ void async_ld_b128(void* lds, const void* g) {
  unsigned l = (unsigned)(uintptr_t)lds;
  unsigned long long ga = (unsigned long long)(uintptr_t)g;
  asm volatile("global_load_async_to_lds_b128 %0, %1, off"
               :: "v"(l), "v"(ga) : "memory");
}
#define WAIT_ASYNC_0() asm volatile("s_wait_asynccnt 0x0" ::: "memory")
#define WAIT_ASYNC_2() asm volatile("s_wait_asynccnt 0x2" ::: "memory")
#define WAIT_DS_0()    asm volatile("s_wait_dscnt 0x0"    ::: "memory")

// ---------------------------------------------------------------------------
// elementwise f32 -> bf16
__global__ void convert_f32_bf16(const float* __restrict__ in,
                                 u16* __restrict__ out, int n) {
  int i = blockIdx.x * blockDim.x + threadIdx.x;
  if (i < n) out[i] = f32_to_bf16(in[i]);
}

// transpose (rows x cols) f32 -> (cols x rows) bf16
__global__ void transpose_f32_bf16(const float* __restrict__ in,
                                   u16* __restrict__ out, int rows, int cols) {
  int i = blockIdx.x * blockDim.x + threadIdx.x;
  if (i < rows * cols) {
    int k = i / cols, n = i % cols;
    out[(size_t)n * rows + k] = f32_to_bf16(in[i]);
  }
}

// ---------------------------------------------------------------------------
// GEMM: C(M x N) = A(M x 1024, bf16) * Bt(N x 1024, bf16)^T + bias
// block = 256 threads = 8 waves; wave computes 32(M) x 64(N); block 128x128.
// mode 0: scatter epilogue to Q (scaled), K, Vt.  mode 1: f32 to Out.
__global__ __launch_bounds__(256) void gemm_bf16_wmma(
    const u16* __restrict__ A, const u16* __restrict__ Bt,
    const float* __restrict__ bias, int mode,
    u16* __restrict__ Qd, u16* __restrict__ Kd, u16* __restrict__ Vtd,
    float* __restrict__ Out) {
  const int lane = threadIdx.x & 31;
  const int wid  = threadIdx.x >> 5;
  const int half = lane >> 4;
  const int lid  = lane & 15;
  const int Mbase = blockIdx.x * 128 + (wid >> 1) * 32;
  const int Nbase = blockIdx.y * 128 + (wid & 1) * 64;
  const int akofs = half * 8;    // A frag: 8+gap+8 K pattern
  const int bkofs = half * 8;    // B frag: 8+gap+8 K pattern

  v8f acc[2][4] = {};

  const u16* arow[2];
  arow[0] = A + (size_t)(Mbase + lid) * 1024;
  arow[1] = A + (size_t)(Mbase + 16 + lid) * 1024;
  const u16* brow[4];
#pragma unroll
  for (int t = 0; t < 4; ++t)
    brow[t] = Bt + (size_t)(Nbase + t * 16 + lid) * 1024;

  for (int k0 = 0; k0 < 1024; k0 += 32) {
    if (k0 + 160 <= 1024) {
      __builtin_prefetch(arow[0] + k0 + 128, 0, 1);
      __builtin_prefetch(brow[0] + k0 + 128, 0, 1);
    }
    BFrag a[2], b[4];
#pragma unroll
    for (int i = 0; i < 2; ++i) {
      a[i].q[0] = *(const uint4*)(arow[i] + k0 + akofs);
      a[i].q[1] = *(const uint4*)(arow[i] + k0 + akofs + 16);
    }
#pragma unroll
    for (int t = 0; t < 4; ++t) {
      b[t].q[0] = *(const uint4*)(brow[t] + k0 + bkofs);
      b[t].q[1] = *(const uint4*)(brow[t] + k0 + bkofs + 16);
    }
#pragma unroll
    for (int i = 0; i < 2; ++i)
#pragma unroll
      for (int t = 0; t < 4; ++t)
        acc[i][t] = __builtin_amdgcn_wmma_f32_16x16x32_bf16(
            false, a[i].v, false, b[t].v, (short)0, acc[i][t], false, false);
  }

#pragma unroll
  for (int i = 0; i < 2; ++i) {
#pragma unroll
    for (int t = 0; t < 4; ++t) {
      const int col = Nbase + t * 16 + lid;
      const float bs = bias[col];
#pragma unroll
      for (int r = 0; r < 8; ++r) {
        const int row = Mbase + i * 16 + r + half * 8;   // C layout: M = r + 8*(lane>=16)
        const float val = acc[i][t][r] + bs;
        if (mode == 1) {
          Out[(size_t)row * 1024 + col] = val;
        } else {
          const int which = col >> 10;
          const int h  = (col & 1023) >> 6;
          const int hd = col & 63;
          const int bb = row >> 11;
          const int tt = row & 2047;
          const int bh = bb * 16 + h;
          if (which == 0)
            Qd[((size_t)bh * 2048 + tt) * 64 + hd] = f32_to_bf16(val * 0.125f); // 1/sqrt(64)
          else if (which == 1)
            Kd[((size_t)bh * 2048 + tt) * 64 + hd] = f32_to_bf16(val);
          else
            Vtd[((size_t)bh * 64 + hd) * 2048 + tt] = f32_to_bf16(val);
        }
      }
    }
  }
}

// ---------------------------------------------------------------------------
// Block-cooperative flash attention: 8 waves (one 16-row q tile each) share
// K/V chunks of 32 keys staged in LDS by double-buffered async copies.
// grid = (T/128, B*H). Q,K: (bh, T, 64) bf16 (Q pre-scaled); Vt: (bh, 64, T).
__global__ __launch_bounds__(256) void attn_flash_wmma(
    const u16* __restrict__ Q, const u16* __restrict__ K,
    const u16* __restrict__ Vt, u16* __restrict__ Yb) {
  __shared__ __align__(16) u16 kbuf[2][32 * 64];  // 32 keys x 64 hd
  __shared__ __align__(16) u16 vbuf[2][64 * 32];  // 64 hd x 32 keys (V^T)
  __shared__ __align__(16) u16 plds[8 * 512];     // 16x32 P tile per wave

  const int tid  = threadIdx.x;
  const int lane = tid & 31;
  const int wid  = tid >> 5;
  const int half = lane >> 4;
  const int lid  = lane & 15;
  const int akofs = half * 8;
  const int bkofs = half * 8;

  const int qt = blockIdx.x * 8 + wid;
  const int qbase = qt * 16;
  const int bh = blockIdx.y;

  const u16* Qh = Q  + (size_t)bh * 2048 * 64;
  const u16* Kh = K  + (size_t)bh * 2048 * 64;
  const u16* Vh = Vt + (size_t)bh * 64 * 2048;
  u16* pw = plds + wid * 512;

  // async staging coordinates: 256 threads x 16B cover one 4KB chunk
  const int krow = tid >> 3, kseg = (tid & 7) * 8;  // K: 32 rows x 128B
  const int vrow = tid >> 2, vseg = (tid & 3) * 8;  // V: 64 rows x  64B

  const int nch  = 4 * (blockIdx.x + 1);      // key chunks for whole block
  const int mych = (qbase + 15) >> 5;         // last chunk this wave needs

  // Q A-fragments (16 x 64 -> two K=32 fragments), held in registers.
  BFrag aq[2];
  const u16* qrow = Qh + (size_t)(qbase + lid) * 64;
#pragma unroll
  for (int j = 0; j < 2; ++j) {
    aq[j].q[0] = *(const uint4*)(qrow + j * 32 + akofs);
    aq[j].q[1] = *(const uint4*)(qrow + j * 32 + akofs + 16);
  }

  v8f yacc[4] = {};
  float mrow[8], lrow[8];
#pragma unroll
  for (int r = 0; r < 8; ++r) { mrow[r] = NEG_BIG; lrow[r] = 0.f; }

  // stage chunk 0
  async_ld_b128(&kbuf[0][krow * 64 + kseg], Kh + (size_t)krow * 64 + kseg);
  async_ld_b128(&vbuf[0][vrow * 32 + vseg], Vh + (size_t)vrow * 2048 + vseg);

  for (int kc = 0; kc < nch; ++kc) {
    const int ks0 = kc * 32;
    const int cur = kc & 1;
    if (kc + 1 < nch) {                       // prefetch next chunk
      const int nxt = cur ^ 1, ksn = ks0 + 32;
      async_ld_b128(&kbuf[nxt][krow * 64 + kseg],
                    Kh + (size_t)(ksn + krow) * 64 + kseg);
      async_ld_b128(&vbuf[nxt][vrow * 32 + vseg],
                    Vh + (size_t)vrow * 2048 + ksn + vseg);
      WAIT_ASYNC_2();                         // chunk kc landed (in-order)
    } else {
      WAIT_ASYNC_0();
    }
    __syncthreads();                          // all waves' copies visible

    if (kc <= mych) {
      const u16* kb = kbuf[cur];
      const u16* vb = vbuf[cur];
#pragma unroll
      for (int sub = 0; sub < 2; ++sub) {
        const int ksub = ks0 + sub * 16;
        // B fragments of K^T from LDS: column = key, contiguous in hd
        BFrag bk[2];
        const u16* krw = kb + (sub * 16 + lid) * 64;
#pragma unroll
        for (int j = 0; j < 2; ++j) {
          bk[j].q[0] = *(const uint4*)(krw + j * 32 + bkofs);
          bk[j].q[1] = *(const uint4*)(krw + j * 32 + bkofs + 16);
        }
        v8f s = {};
        s = __builtin_amdgcn_wmma_f32_16x16x32_bf16(false, aq[0].v, false, bk[0].v,
                                                    (short)0, s, false, false);
        s = __builtin_amdgcn_wmma_f32_16x16x32_bf16(false, aq[1].v, false, bk[1].v,
                                                    (short)0, s, false, false);
        // online softmax in C-layout: row m = r + 8*half, col n = lid
#pragma unroll
        for (int r = 0; r < 8; ++r) {
          const int m = r + half * 8;
          float sv = s[r];
          if (ksub + lid > qbase + m) sv = NEG_BIG;   // causal mask
          float rmax = sv;
#pragma unroll
          for (int off = 1; off < 16; off <<= 1)
            rmax = fmaxf(rmax, __shfl_xor(rmax, off, 32));
          const float nm = fmaxf(mrow[r], rmax);
          const float alpha = __expf(mrow[r] - nm);
          const float p = __expf(sv - nm);
          float rsum = p;
#pragma unroll
          for (int off = 1; off < 16; off <<= 1)
            rsum += __shfl_xor(rsum, off, 32);
          lrow[r] = lrow[r] * alpha + rsum;
          mrow[r] = nm;
#pragma unroll
          for (int t = 0; t < 4; ++t) yacc[t][r] *= alpha;
          pw[m * 32 + sub * 16 + lid] = f32_to_bf16(p);
        }
      }
      WAIT_DS_0();                            // P tile visible across lanes
      // P (16x32) A-fragment from LDS
      BFrag pf;
      pf.q[0] = *(const uint4*)(pw + lid * 32 + akofs);
      pf.q[1] = *(const uint4*)(pw + lid * 32 + akofs + 16);
      // y += P @ V(32 x 64): B columns = hd rows of V^T chunk in LDS
#pragma unroll
      for (int t = 0; t < 4; ++t) {
        BFrag bv;
        const u16* vrw = vb + (t * 16 + lid) * 32;
        bv.q[0] = *(const uint4*)(vrw + bkofs);
        bv.q[1] = *(const uint4*)(vrw + bkofs + 16);
        yacc[t] = __builtin_amdgcn_wmma_f32_16x16x32_bf16(
            false, pf.v, false, bv.v, (short)0, yacc[t], false, false);
      }
    }
    __syncthreads();                          // protect buffers for next stage
  }

  // normalize and store to (B, T, C) bf16
  const int bb = bh >> 4, hh = bh & 15;
#pragma unroll
  for (int r = 0; r < 8; ++r) {
    const int m = r + half * 8;
    const int tg = qbase + m;
    const float inv = 1.0f / lrow[r];
#pragma unroll
    for (int t = 0; t < 4; ++t)
      Yb[(size_t)(bb * 2048 + tg) * 1024 + hh * 64 + t * 16 + lid] =
          f32_to_bf16(yacc[t][r] * inv);
  }
}

// ---------------------------------------------------------------------------
extern "C" void kernel_launch(void* const* d_in, const int* in_sizes, int n_in,
                              void* d_out, int out_size, void* d_ws, size_t ws_size,
                              hipStream_t stream) {
  const float* x     = (const float*)d_in[0];   // (4,2048,1024)
  const float* W_qkv = (const float*)d_in[1];   // (1024,3072)
  const float* b_qkv = (const float*)d_in[2];   // (3072)
  const float* W_out = (const float*)d_in[3];   // (1024,1024)
  const float* b_out = (const float*)d_in[4];   // (1024)
  float* out = (float*)d_out;                   // (4,2048,1024) f32

  char* ws = (char*)d_ws;
  u16* xb    = (u16*)ws; ws += (size_t)8192 * 1024 * 2;  // x bf16
  u16* wqkvt = (u16*)ws; ws += (size_t)3072 * 1024 * 2;  // W_qkv^T bf16
  u16* woutt = (u16*)ws; ws += (size_t)1024 * 1024 * 2;  // W_out^T bf16
  u16* Qd    = (u16*)ws; ws += (size_t)64 * 2048 * 64 * 2;
  u16* Kd    = (u16*)ws; ws += (size_t)64 * 2048 * 64 * 2;
  u16* Vtd   = (u16*)ws; ws += (size_t)64 * 64 * 2048 * 2;
  u16* Yb    = (u16*)ws; ws += (size_t)8192 * 1024 * 2;  // attention out bf16

  {
    int n = 8192 * 1024;
    convert_f32_bf16<<<(n + 255) / 256, 256, 0, stream>>>(x, xb, n);
  }
  {
    int n = 1024 * 3072;
    transpose_f32_bf16<<<(n + 255) / 256, 256, 0, stream>>>(W_qkv, wqkvt, 1024, 3072);
  }
  {
    int n = 1024 * 1024;
    transpose_f32_bf16<<<(n + 255) / 256, 256, 0, stream>>>(W_out, woutt, 1024, 1024);
  }
  // QKV projection: 8192 x 3072, scatter into Q(scaled)/K/Vt
  gemm_bf16_wmma<<<dim3(64, 24), 256, 0, stream>>>(
      xb, wqkvt, b_qkv, 0, Qd, Kd, Vtd, nullptr);
  // causal flash attention (async-LDS staged K/V)
  attn_flash_wmma<<<dim3(16, 64), 256, 0, stream>>>(Qd, Kd, Vtd, Yb);
  // output projection: 8192 x 1024 -> f32
  gemm_bf16_wmma<<<dim3(64, 8), 256, 0, stream>>>(
      Yb, woutt, b_out, 1, nullptr, nullptr, nullptr, out);
}
